// Attention_2H_GRU_80092550136357
// MI455X (gfx1250) — matrix operand order repaired and checked by reading, not compiled, not run on hardware
//
#include <hip/hip_runtime.h>
#include <hip/hip_bf16.h>
#include <math.h>

// Problem constants (match reference)
#define B_   32
#define T_   128
#define S_   32
#define W_   64
#define U_   256
#define D_   256
#define BS_  (B_*S_)       // 1024
#define BSW_ (B_*S_*W_)    // 65536

typedef __attribute__((ext_vector_type(16))) _Float16 v16h;
typedef __attribute__((ext_vector_type(8)))  _Float16 v8h;
typedef __attribute__((ext_vector_type(8)))  float    v8f;

__device__ __forceinline__ float hsig(float x) {
    return fminf(fmaxf(0.2f * x + 0.5f, 0.0f), 1.0f);
}

// ---------------------------------------------------------------------------
// f32 -> f16 flat convert
__global__ void cvt_kernel(const float* __restrict__ in, _Float16* __restrict__ out, int n) {
    int i = blockIdx.x * blockDim.x + threadIdx.x;
    if (i < n) out[i] = (_Float16)in[i];
}

// f32 [rows x Ntot] sub-block (rows r0..r0+Krows, cols c0..c0+N) -> f16 transposed [N x Krows]
__global__ void cvt_t_kernel(const float* __restrict__ in, _Float16* __restrict__ out,
                             int Krows, int Ntot, int r0, int c0, int N) {
    int i = blockIdx.x * blockDim.x + threadIdx.x;
    if (i >= N * Krows) return;
    int n = i / Krows;
    int k = i - n * Krows;
    out[i] = (_Float16)in[(size_t)(r0 + k) * Ntot + c0 + n];
}

// ---------------------------------------------------------------------------
// Fragment loaders, contiguous k order k = 16*grp + s (two contiguous b128 loads each).
__device__ __forceinline__ v16h load_frag_a(const _Float16* __restrict__ base) {
    // base = A + row*K + k0 + grp*16 ; elements 0..15 -> contiguous K
    v8h lo = *(const v8h*)(base);
    v8h hi = *(const v8h*)(base + 8);
    v16h f;
#pragma unroll
    for (int i = 0; i < 8; ++i) { f[i] = lo[i]; f[i + 8] = hi[i]; }
    return f;
}
__device__ __forceinline__ v16h load_frag_b(const _Float16* __restrict__ base) {
    // base = Bt + col*K + k0 + grp*16 ; elements 0..15 -> contiguous K
    v8h lo = *(const v8h*)(base);
    v8h hi = *(const v8h*)(base + 8);
    v16h f;
#pragma unroll
    for (int i = 0; i < 8; ++i) { f[i] = lo[i]; f[i + 8] = hi[i]; }
    return f;
}

// Per-16x16-subtile epilogue.
// MODE 0: store f32 (+ optional bias[col])
// MODE 1: store f16 (+ optional bias[col])
// MODE 2: fused score: atomicAdd(e[row], sum_col tanh(acc+bias[col])*k3[col])
template<int MODE>
__device__ __forceinline__ void epilogue_tile(v8f acc, int rowbase, int colbase, int lane,
                                              float* __restrict__ Cf, _Float16* __restrict__ Ch,
                                              const float* __restrict__ bias,
                                              const float* __restrict__ k3,
                                              float* __restrict__ escore, int N)
{
    const int r16 = lane & 15;
    const int grp = lane >> 4;
    const int col = colbase + r16;
    if (MODE == 2) {
#pragma unroll
        for (int r = 0; r < 8; ++r) {
            int row = rowbase + r + grp * 8;
            float v = tanhf(acc[r] + bias[col]) * k3[col];
            v += __shfl_xor(v, 1);
            v += __shfl_xor(v, 2);
            v += __shfl_xor(v, 4);
            v += __shfl_xor(v, 8);
            if (r16 == 0) atomicAdd(&escore[row], v);
        }
    } else {
#pragma unroll
        for (int r = 0; r < 8; ++r) {
            int row = rowbase + r + grp * 8;
            float v = acc[r] + (bias ? bias[col] : 0.0f);
            if (MODE == 0) Cf[(size_t)row * N + col] = v;
            else           Ch[(size_t)row * N + col] = (_Float16)v;
        }
    }
}

// ---------------------------------------------------------------------------
// WMMA GEMM: C[M,N] = A[M,K] (f16, row-major) x Bt[N,K] (f16, pre-transposed).
// One wave computes one 32x32 output super-tile: 2x2 register blocking,
// 4 v_wmma_f32_16x16x32_f16 per k-step from 2 A + 2 B fragments (16 FLOP/B),
// with 4 independent accumulator chains to hide WMMA latency.
// Requires M % 32 == 0, N % 32 == 0, K % 32 == 0.
template<int MODE>
__global__ __launch_bounds__(128)
void gemm_wmma(const _Float16* __restrict__ A, const _Float16* __restrict__ Bt,
               float* __restrict__ Cf, _Float16* __restrict__ Ch,
               const float* __restrict__ bias,
               const float* __restrict__ k3, float* __restrict__ escore,
               int M, int N, int K)
{
    const int lane = threadIdx.x & 31;
    const int wave = threadIdx.x >> 5;
    const int nsup = N >> 5;
    const int msup = M >> 5;
    const int tile = blockIdx.x * 4 + wave;
    if (tile >= msup * nsup) return;            // wave-uniform exit
    const int tm = tile / nsup;
    const int tn = tile - tm * nsup;

    const int r16 = lane & 15;
    const int grp = lane >> 4;
    const size_t arow0 = (size_t)(tm * 32 + r16) * K;
    const size_t arow1 = arow0 + (size_t)16 * K;
    const size_t brow0 = (size_t)(tn * 32 + r16) * K;
    const size_t brow1 = brow0 + (size_t)16 * K;

    v8f acc00 = {}, acc01 = {}, acc10 = {}, acc11 = {};
    for (int k0 = 0; k0 < K; k0 += 32) {
        v16h a0 = load_frag_a(A + arow0 + k0 + grp * 16);
        v16h a1 = load_frag_a(A + arow1 + k0 + grp * 16);
        v16h b0 = load_frag_b(Bt + brow0 + k0 + grp * 16);
        v16h b1 = load_frag_b(Bt + brow1 + k0 + grp * 16);
        acc00 = __builtin_amdgcn_wmma_f32_16x16x32_f16(false, a0, false, b0, (short)0, acc00, false, false);
        acc01 = __builtin_amdgcn_wmma_f32_16x16x32_f16(false, a0, false, b1, (short)0, acc01, false, false);
        acc10 = __builtin_amdgcn_wmma_f32_16x16x32_f16(false, a1, false, b0, (short)0, acc10, false, false);
        acc11 = __builtin_amdgcn_wmma_f32_16x16x32_f16(false, a1, false, b1, (short)0, acc11, false, false);
    }

    epilogue_tile<MODE>(acc00, tm * 32,      tn * 32,      lane, Cf, Ch, bias, k3, escore, N);
    epilogue_tile<MODE>(acc01, tm * 32,      tn * 32 + 16, lane, Cf, Ch, bias, k3, escore, N);
    epilogue_tile<MODE>(acc10, tm * 32 + 16, tn * 32,      lane, Cf, Ch, bias, k3, escore, N);
    epilogue_tile<MODE>(acc11, tm * 32 + 16, tn * 32 + 16, lane, Cf, Ch, bias, k3, escore, N);
}

// ---------------------------------------------------------------------------
// act1 = tanh(hproj1[b] + wproj + a1b1), f16 out.  rows = B*S*W, hproj stride 512.
__global__ void act_word_kernel(const float* __restrict__ hproj12,
                                const _Float16* __restrict__ wproj,
                                const float* __restrict__ b1,
                                _Float16* __restrict__ out)
{
    int i = blockIdx.x * blockDim.x + threadIdx.x;
    if (i >= BSW_ * U_) return;
    int d   = i & (U_ - 1);
    int row = i >> 8;
    int b   = row >> 11;                        // row / (S*W)
    float v = hproj12[b * 512 + d] + (float)wproj[i] + b1[d];
    out[i] = (_Float16)tanhf(v);
}

// act2 = tanh(hproj2[b] + sproj + a2b1), f16 out.  rows = B*S.
__global__ void act_sent_kernel(const float* __restrict__ hproj12,
                                const float* __restrict__ sproj,
                                const float* __restrict__ b1,
                                _Float16* __restrict__ out)
{
    int i = blockIdx.x * blockDim.x + threadIdx.x;
    if (i >= BS_ * U_) return;
    int d   = i & (U_ - 1);
    int row = i >> 8;
    int b   = row >> 5;                         // row / S
    float v = hproj12[b * 512 + 256 + d] + sproj[i] + b1[d];
    out[i] = (_Float16)tanhf(v);
}

// e_w[i] = a1b3, e_s[i] = a2b3 (score bias init before fused-GEMM atomic accumulation)
__global__ void step_init_kernel(float* __restrict__ e_w, float* __restrict__ e_s,
                                 const float* __restrict__ b3w, const float* __restrict__ b3s)
{
    int i = blockIdx.x * blockDim.x + threadIdx.x;
    if (i < BSW_) e_w[i] = b3w[0];
    if (i < BS_)  e_s[i] = b3s[0];
}

// word softmax over W=64 + weighted context: wctx[bs,d] = sum_w alpha*att_w
__global__ void softmax_word_kernel(const float* __restrict__ e_w,
                                    const float* __restrict__ att_w,
                                    float* __restrict__ wctx)
{
    __shared__ float sh[W_];
    int bs = blockIdx.x;                        // 0..B*S-1
    int d  = threadIdx.x;                       // 0..255
    if (d < W_) sh[d] = e_w[bs * W_ + d];
    __syncthreads();
    float m = -1e30f;
    for (int w = 0; w < W_; ++w) m = fmaxf(m, sh[w]);
    float ssum = 0.0f;
    for (int w = 0; w < W_; ++w) ssum += __expf(sh[w] - m);
    float inv = 1.0f / ssum;
    float acc = 0.0f;
    for (int w = 0; w < W_; ++w)
        acc += __expf(sh[w] - m) * inv * att_w[((size_t)bs * W_ + w) * U_ + d];
    wctx[(size_t)bs * U_ + d] = acc;
}

// sentence softmax over S=32 + context: c[b,d] = sum_s beta*wctx
__global__ void softmax_sent_kernel(const float* __restrict__ e_s,
                                    const float* __restrict__ wctx,
                                    float* __restrict__ c, _Float16* __restrict__ c_h)
{
    __shared__ float sh[S_];
    int b = blockIdx.x;
    int d = threadIdx.x;
    if (d < S_) sh[d] = e_s[b * S_ + d];
    __syncthreads();
    float m = -1e30f;
    for (int s = 0; s < S_; ++s) m = fmaxf(m, sh[s]);
    float ssum = 0.0f;
    for (int s = 0; s < S_; ++s) ssum += __expf(sh[s] - m);
    float inv = 1.0f / ssum;
    float acc = 0.0f;
    for (int s = 0; s < S_; ++s)
        acc += __expf(sh[s] - m) * inv * wctx[((size_t)b * S_ + s) * U_ + d];
    c[b * U_ + d]   = acc;
    c_h[b * U_ + d] = (_Float16)acc;
}

// GRU gates z,r; rh = r*h (f16)
__global__ void gate1_kernel(const float* __restrict__ xproj, const float* __restrict__ hzr,
                             const float* __restrict__ call, const float* __restrict__ h,
                             float* __restrict__ z, _Float16* __restrict__ rh, int t)
{
    int i = blockIdx.x * blockDim.x + threadIdx.x;
    if (i >= B_ * U_) return;
    int b = i >> 8, d = i & (U_ - 1);
    const float* xt = xproj + ((size_t)b * T_ + t) * (3 * U_);
    float zv = hsig(xt[d]      + hzr[b * 512 + d]      + call[b * 768 + d]);
    float rv = hsig(xt[U_ + d] + hzr[b * 512 + U_ + d] + call[b * 768 + U_ + d]);
    z[i]  = zv;
    rh[i] = (_Float16)(rv * h[i]);
}

// hh = tanh(...); h_new = z*h + (1-z)*hh; writes h (f32+f16) and output slice
__global__ void gate2_kernel(const float* __restrict__ xproj, const float* __restrict__ rhW,
                             const float* __restrict__ call, const float* __restrict__ z,
                             float* __restrict__ h, _Float16* __restrict__ h_h,
                             float* __restrict__ out, int t)
{
    int i = blockIdx.x * blockDim.x + threadIdx.x;
    if (i >= B_ * U_) return;
    int b = i >> 8, d = i & (U_ - 1);
    const float* xt = xproj + ((size_t)b * T_ + t) * (3 * U_);
    float hh = tanhf(xt[2 * U_ + d] + rhW[i] + call[b * 768 + 2 * U_ + d]);
    float hn = z[i] * h[i] + (1.0f - z[i]) * hh;
    h[i]   = hn;
    h_h[i] = (_Float16)hn;
    out[((size_t)b * T_ + t) * U_ + d] = hn;
}

__global__ void zero_h_kernel(float* __restrict__ h, _Float16* __restrict__ h_h) {
    int i = blockIdx.x * blockDim.x + threadIdx.x;
    if (i < B_ * U_) { h[i] = 0.0f; h_h[i] = (_Float16)0.0f; }
}

// ---------------------------------------------------------------------------
extern "C" void kernel_launch(void* const* d_in, const int* in_sizes, int n_in,
                              void* d_out, int out_size, void* d_ws, size_t ws_size,
                              hipStream_t stream)
{
    (void)in_sizes; (void)n_in; (void)out_size; (void)ws_size;
    const float* x     = (const float*)d_in[0];
    const float* att_s = (const float*)d_in[1];
    const float* att_w = (const float*)d_in[2];
    const float* W_xA  = (const float*)d_in[3];
    const float* W_hA  = (const float*)d_in[4];
    const float* W_cA  = (const float*)d_in[5];
    const float* bias  = (const float*)d_in[6];
    const float* a1k1  = (const float*)d_in[7];
    const float* a1k2  = (const float*)d_in[8];
    const float* a1k3  = (const float*)d_in[9];
    const float* a1b1  = (const float*)d_in[10];
    const float* a1b2  = (const float*)d_in[11];
    const float* a1b3  = (const float*)d_in[12];
    const float* a2k1  = (const float*)d_in[13];
    const float* a2k2  = (const float*)d_in[14];
    const float* a2k3  = (const float*)d_in[15];
    const float* a2b1  = (const float*)d_in[16];
    const float* a2b2  = (const float*)d_in[17];
    const float* a2b3  = (const float*)d_in[18];
    float* out = (float*)d_out;

    // --- workspace bump allocator (256B aligned); total ~122 MB ---
    char* p = (char*)d_ws;
    auto alloc = [&](size_t bytes) -> void* {
        void* r = (void*)p;
        p += (bytes + 255) & ~(size_t)255;
        return r;
    };
    _Float16* WxA_t    = (_Float16*)alloc((size_t)768 * 256 * 2);  // [3U, D]
    _Float16* k1topT   = (_Float16*)alloc((size_t)512 * 256 * 2);  // concat a1k1_top^T | a2k1_top^T
    _Float16* a1k1botT = (_Float16*)alloc((size_t)256 * 256 * 2);
    _Float16* a2k1botT = (_Float16*)alloc((size_t)256 * 256 * 2);
    _Float16* a1k2T    = (_Float16*)alloc((size_t)256 * 256 * 2);
    _Float16* a2k2T    = (_Float16*)alloc((size_t)256 * 256 * 2);
    _Float16* WhA_zrT  = (_Float16*)alloc((size_t)512 * 256 * 2);
    _Float16* WhA_hT   = (_Float16*)alloc((size_t)256 * 256 * 2);
    _Float16* WcA_T    = (_Float16*)alloc((size_t)768 * 256 * 2);
    _Float16* x_h      = (_Float16*)alloc((size_t)B_ * T_ * D_ * 2);
    _Float16* att_w_h  = (_Float16*)alloc((size_t)BSW_ * U_ * 2);
    _Float16* att_s_h  = (_Float16*)alloc((size_t)BS_ * U_ * 2);
    float*    xproj    = (float*)   alloc((size_t)B_ * T_ * 3 * U_ * 4);
    _Float16* wproj    = (_Float16*)alloc((size_t)BSW_ * U_ * 2);
    float*    sproj    = (float*)   alloc((size_t)BS_ * U_ * 4);
    _Float16* act1     = (_Float16*)alloc((size_t)BSW_ * U_ * 2);
    _Float16* act2     = (_Float16*)alloc((size_t)BS_ * U_ * 2);
    float*    e_w      = (float*)   alloc((size_t)BSW_ * 4);
    float*    e_s      = (float*)   alloc((size_t)BS_ * 4);
    float*    wctx     = (float*)   alloc((size_t)BS_ * U_ * 4);
    float*    cvec     = (float*)   alloc((size_t)B_ * U_ * 4);
    _Float16* cvec_h   = (_Float16*)alloc((size_t)B_ * U_ * 2);
    float*    h        = (float*)   alloc((size_t)B_ * U_ * 4);
    _Float16* h_h      = (_Float16*)alloc((size_t)B_ * U_ * 2);
    float*    hproj12  = (float*)   alloc((size_t)B_ * 512 * 4);
    float*    hzr      = (float*)   alloc((size_t)B_ * 512 * 4);
    float*    call_    = (float*)   alloc((size_t)B_ * 768 * 4);
    float*    zbuf     = (float*)   alloc((size_t)B_ * U_ * 4);
    _Float16* rh       = (_Float16*)alloc((size_t)B_ * U_ * 2);
    float*    rhW      = (float*)   alloc((size_t)B_ * U_ * 4);

    auto cdiv = [](size_t a, size_t b) { return (unsigned)((a + b - 1) / b); };
    // blocks for a GEMM of given M,N with 32x32 tiles, 4 waves/block
    auto gblocks = [&](int M, int N) { return cdiv((size_t)(M / 32) * (N / 32), 4); };

    // --- weight transpose+convert (once) ---
    // out[n*K+k] = in[(r0+k)*Ntot + c0+n]
    cvt_t_kernel<<<cdiv(768 * 256, 256), 256, 0, stream>>>(W_xA, WxA_t, 256, 768, 0, 0, 768);
    cvt_t_kernel<<<cdiv(256 * 256, 256), 256, 0, stream>>>(a1k1, k1topT, 256, 256, 0, 0, 256);
    cvt_t_kernel<<<cdiv(256 * 256, 256), 256, 0, stream>>>(a2k1, k1topT + 256 * 256, 256, 256, 0, 0, 256);
    cvt_t_kernel<<<cdiv(256 * 256, 256), 256, 0, stream>>>(a1k1, a1k1botT, 256, 256, 256, 0, 256);
    cvt_t_kernel<<<cdiv(256 * 256, 256), 256, 0, stream>>>(a2k1, a2k1botT, 256, 256, 256, 0, 256);
    cvt_t_kernel<<<cdiv(256 * 256, 256), 256, 0, stream>>>(a1k2, a1k2T, 256, 256, 0, 0, 256);
    cvt_t_kernel<<<cdiv(256 * 256, 256), 256, 0, stream>>>(a2k2, a2k2T, 256, 256, 0, 0, 256);
    cvt_t_kernel<<<cdiv(512 * 256, 256), 256, 0, stream>>>(W_hA, WhA_zrT, 256, 768, 0, 0, 512);
    cvt_t_kernel<<<cdiv(256 * 256, 256), 256, 0, stream>>>(W_hA, WhA_hT, 256, 768, 0, 512, 256);
    cvt_t_kernel<<<cdiv(768 * 256, 256), 256, 0, stream>>>(W_cA, WcA_T, 256, 768, 0, 0, 768);

    // --- input converts (once) ---
    cvt_kernel<<<cdiv((size_t)B_ * T_ * D_, 256), 256, 0, stream>>>(x, x_h, B_ * T_ * D_);
    cvt_kernel<<<cdiv((size_t)BSW_ * U_, 256), 256, 0, stream>>>(att_w, att_w_h, BSW_ * U_);
    cvt_kernel<<<cdiv((size_t)BS_ * U_, 256), 256, 0, stream>>>(att_s, att_s_h, BS_ * U_);

    // --- step-invariant GEMMs ---
    // xproj = x @ W_xA + bias : [4096,768]
    gemm_wmma<0><<<gblocks(B_ * T_, 768), 128, 0, stream>>>(
        x_h, WxA_t, xproj, nullptr, bias, nullptr, nullptr, B_ * T_, 768, 256);
    // wproj = att_w @ a1k1_bot : [65536,256] (f16)
    gemm_wmma<1><<<gblocks(BSW_, U_), 128, 0, stream>>>(
        att_w_h, a1k1botT, nullptr, wproj, nullptr, nullptr, nullptr, BSW_, U_, 256);
    // sproj = att_s @ a2k1_bot : [1024,256]
    gemm_wmma<0><<<gblocks(BS_, U_), 128, 0, stream>>>(
        att_s_h, a2k1botT, sproj, nullptr, nullptr, nullptr, nullptr, BS_, U_, 256);

    zero_h_kernel<<<cdiv((size_t)B_ * U_, 256), 256, 0, stream>>>(h, h_h);

    // --- recurrence ---
    for (int t = 0; t < T_; ++t) {
        // hproj12 = h @ [a1k1_top | a2k1_top] : [32,512]
        gemm_wmma<0><<<gblocks(B_, 512), 128, 0, stream>>>(
            h_h, k1topT, hproj12, nullptr, nullptr, nullptr, nullptr, B_, 512, 256);
        act_word_kernel<<<cdiv((size_t)BSW_ * U_, 256), 256, 0, stream>>>(hproj12, wproj, a1b1, act1);
        act_sent_kernel<<<cdiv((size_t)BS_ * U_, 256), 256, 0, stream>>>(hproj12, sproj, a2b1, act2);
        step_init_kernel<<<cdiv((size_t)BSW_, 256), 256, 0, stream>>>(e_w, e_s, a1b3, a2b3);
        // fused layer-2 word: e_w += tanh(act1@a1k2 + a1b2) . a1k3   (dominant GEMM)
        gemm_wmma<2><<<gblocks(BSW_, U_), 128, 0, stream>>>(
            act1, a1k2T, nullptr, nullptr, a1b2, a1k3, e_w, BSW_, U_, 256);
        // fused layer-2 sentence
        gemm_wmma<2><<<gblocks(BS_, U_), 128, 0, stream>>>(
            act2, a2k2T, nullptr, nullptr, a2b2, a2k3, e_s, BS_, U_, 256);
        softmax_word_kernel<<<BS_, 256, 0, stream>>>(e_w, att_w, wctx);
        softmax_sent_kernel<<<B_, 256, 0, stream>>>(e_s, wctx, cvec, cvec_h);
        // GRU gate GEMMs
        gemm_wmma<0><<<gblocks(B_, 512), 128, 0, stream>>>(
            h_h, WhA_zrT, hzr, nullptr, nullptr, nullptr, nullptr, B_, 512, 256);
        gemm_wmma<0><<<gblocks(B_, 768), 128, 0, stream>>>(
            cvec_h, WcA_T, call_, nullptr, nullptr, nullptr, nullptr, B_, 768, 256);
        gate1_kernel<<<cdiv((size_t)B_ * U_, 256), 256, 0, stream>>>(xproj, hzr, call_, h, zbuf, rh, t);
        gemm_wmma<0><<<gblocks(B_, U_), 128, 0, stream>>>(
            rh, WhA_hT, rhW, nullptr, nullptr, nullptr, nullptr, B_, U_, 256);
        gate2_kernel<<<cdiv((size_t)B_ * U_, 256), 256, 0, stream>>>(xproj, rhW, call_, zbuf, h, h_h, out, t);
    }
}
